// Attention_50921132261730
// MI455X (gfx1250) — matrix operand order repaired and checked by reading, not compiled, not run on hardware
//
#include <hip/hip_runtime.h>
#include <hip/hip_bf16.h>

// Problem constants (from reference)
#define B_  2
#define C_  512
#define N_  2048
#define H_  16
#define D_  64
#define HD_ 1024
#define M3_ 3072   // 3*HD_

typedef __attribute__((ext_vector_type(16))) __bf16 v16bf;
typedef __attribute__((ext_vector_type(8)))  __bf16 v8bf;
typedef __attribute__((ext_vector_type(8)))  float  v8f;
typedef unsigned short u16;

__device__ __forceinline__ u16 f2bf(float f) {
  __bf16 h = (__bf16)f;
  return __builtin_bit_cast(u16, h);
}

__device__ __forceinline__ v8f wmma_bf16(v16bf a, v16bf b, v8f c) {
  // v_wmma_f32_16x16x32_bf16 : D = A(16x32) * B(32x16) + C(16x16 f32)
  return __builtin_amdgcn_wmma_f32_16x16x32_bf16(false, a, false, b, (short)0, c,
                                                 false, false);
}

// A-fragment (contiguous k order): caller passes row base + (hi?16).
// lane<16: elems 0-15 = K{base..base+15}
__device__ __forceinline__ v16bf load_afrag(const u16* p) {
  v8bf lo = *(const v8bf*)(p);
  v8bf hi = *(const v8bf*)(p + 8);
  return __builtin_shufflevector(lo, hi, 0,1,2,3,4,5,6,7,8,9,10,11,12,13,14,15);
}

// ---- CDNA5 async memory->LDS copy (ASYNCcnt-tracked), inline asm path ----
__device__ __forceinline__ unsigned lds_off(const void* p) {
  // flat LDS address: low 32 bits carry the LDS byte offset
  return (unsigned)(unsigned long long)p;
}
__device__ __forceinline__ void async_ld_b128(unsigned ldsoff, const void* g) {
  asm volatile("global_load_async_to_lds_b128 %0, %1, off"
               :: "v"(ldsoff), "v"(g) : "memory");
}
__device__ __forceinline__ void wait_async_le(void) {  // <=4 outstanding (next buffer)
  asm volatile("s_wait_asynccnt 0x4" ::: "memory");
}
__device__ __forceinline__ void wait_async_0(void) {
  asm volatile("s_wait_asynccnt 0x0" ::: "memory");
}

// ---------------- elementwise f32 -> bf16 (weights) ----------------
__global__ void k_cvt_bf16(const float* __restrict__ in, u16* __restrict__ out, int n) {
  int i = blockIdx.x * blockDim.x + threadIdx.x;
  if (i < n) out[i] = f2bf(in[i]);
}

// ---------------- x[b][c][n] f32 -> xT[b][n][c] bf16 ----------------
__global__ void k_transpose_x(const float* __restrict__ x, u16* __restrict__ xT) {
  __shared__ float t[32][33];
  int bid = blockIdx.x;
  int nt = bid % (N_/32);
  int ct = (bid / (N_/32)) % (C_/32);
  int b  = bid / ((N_/32) * (C_/32));
  int tid = threadIdx.x, ln = tid & 31;
  for (int r = tid >> 5; r < 32; r += 8)
    t[r][ln] = x[((size_t)b*C_ + ct*32 + r)*N_ + nt*32 + ln];
  __syncthreads();
  for (int r = tid >> 5; r < 32; r += 8)
    xT[((size_t)b*N_ + nt*32 + r)*C_ + ct*32 + ln] = f2bf(t[ln][r]);
}

// ---------------- QKV GEMM: qkv[b][o][n] = Wqkv[o][c] * x[c][n] ----------------
__global__ void k_gemm_qkv(const u16* __restrict__ A, const u16* __restrict__ Bx,
                           float* __restrict__ qkv) {
  const int lane = threadIdx.x & 31;
  const int warp = threadIdx.x >> 5;
  const int l16 = lane & 15, hi = lane >> 4;
  int w = blockIdx.x * 4 + warp;
  const int NS = N_ / 64;   // 32
  const int MT = M3_ / 16;  // 192
  int b  = w / (MT * NS);
  int r0 = w % (MT * NS);
  int ot = r0 / NS;
  int ns = r0 % NS;
  const u16* arow = A + (size_t)(ot*16 + l16)*C_ + hi*16;
  const u16* brow = Bx + ((size_t)b*N_ + ns*64)*C_ + hi*16;
  v8f a0 = {}, a1 = {}, a2 = {}, a3 = {};
  for (int k0 = 0; k0 < C_; k0 += 32) {
    v16bf a  = load_afrag(arow + k0);
    v16bf b0 = *(const v16bf*)(brow + (size_t)( 0 + l16)*C_ + k0);
    v16bf b1 = *(const v16bf*)(brow + (size_t)(16 + l16)*C_ + k0);
    v16bf b2 = *(const v16bf*)(brow + (size_t)(32 + l16)*C_ + k0);
    v16bf b3 = *(const v16bf*)(brow + (size_t)(48 + l16)*C_ + k0);
    a0 = wmma_bf16(a, b0, a0);
    a1 = wmma_bf16(a, b1, a1);
    a2 = wmma_bf16(a, b2, a2);
    a3 = wmma_bf16(a, b3, a3);
  }
  float* orow = qkv + ((size_t)b*M3_ + ot*16 + hi*8)*N_ + ns*64 + l16;
#pragma unroll
  for (int r = 0; r < 8; ++r) {
    orow[(size_t)r*N_ +  0] = a0[r];
    orow[(size_t)r*N_ + 16] = a1[r];
    orow[(size_t)r*N_ + 32] = a2[r];
    orow[(size_t)r*N_ + 48] = a3[r];
  }
}

// ---------------- L2-normalize q,k over D, apply scales (q also *8) ----------------
__global__ void k_normqk(const float* __restrict__ qkv, const float* __restrict__ qs,
                         const float* __restrict__ ks, u16* __restrict__ qo,
                         u16* __restrict__ ko) {
  __shared__ float t[64][33];
  __shared__ float rn[32];
  int bid = blockIdx.x;
  const int NT = N_ / 32;  // 64
  int nt = bid % NT;
  int h  = (bid / NT) % H_;
  int which = (bid / (NT * H_)) % 2;
  int b  = bid / (NT * H_ * 2);
  int tid = threadIdx.x, ln = tid & 31;
  const float* src = qkv + ((size_t)b*M3_ + which*HD_ + h*D_)*N_ + nt*32;
  for (int d = tid >> 5; d < 64; d += 8)
    t[d][ln] = src[(size_t)d*N_ + ln];
  __syncthreads();
  if (tid < 32) {
    float s = 0.f;
#pragma unroll
    for (int d = 0; d < 64; ++d) { float xv = t[d][tid]; s += xv * xv; }
    rn[tid] = 1.0f / fmaxf(sqrtf(s), 1e-12f);
  }
  __syncthreads();
  const float* sc = which ? ks : qs;
  float mul = which ? 1.0f : 8.0f;  // fold SCALE=8 into q
  u16* dst = (which ? ko : qo) + (((size_t)b*H_ + h)*N_ + nt*32)*D_;
  for (int idx = tid; idx < 32 * 64; idx += 256) {
    int n = idx >> 6, d = idx & 63;
    dst[(size_t)n*D_ + d] = f2bf(t[d][n] * rn[n] * sc[d] * mul);
  }
}

// ---------------- v: qkv[b][2048+od][n] f32 -> vT[b][od][n] bf16 ----------------
__global__ void k_cvt_v(const float* __restrict__ qkv, u16* __restrict__ vo) {
  size_t i = (size_t)blockIdx.x * blockDim.x + threadIdx.x;
  size_t n  = i % N_;
  size_t od = (i / N_) % HD_;
  size_t b  = i / ((size_t)N_ * HD_);
  vo[(b*HD_ + od)*N_ + n] = f2bf(qkv[(b*M3_ + 2*HD_ + od)*N_ + n]);
}

// ---------------- flash attention: softmax(q k^T) v ----------------
// Block = 4 waves sharing one (b,h); K/V j-tiles staged in LDS via async
// copies (double-buffered, ASYNCcnt). Each wave owns one 16-row i-tile.
// q,k: [B][H][N][64] bf16 (q pre-scaled), v: [B][H][64][N] bf16
// ao:  [B][N][HD] bf16 (B-matrix layout for the final GEMM)
__global__ void k_flash(const u16* __restrict__ q, const u16* __restrict__ kk,
                        const u16* __restrict__ vv, u16* __restrict__ ao) {
  __shared__ u16 Kl[2][32 * 64];   // 2 x 4KB : K tile, rows j(32) x d(64)
  __shared__ u16 Vl[2][64 * 32];   // 2 x 4KB : V tile, rows d(64) x j(32)
  __shared__ u16 Pl[4][16 * 32];   // per-wave P staging (C->A relayout)
  const int tid  = threadIdx.x;
  const int lane = tid & 31;
  const int warp = tid >> 5;
  const int l16 = lane & 15, hi = lane >> 4;

  const int IT4 = N_ / 64;  // 32 i-tile groups per (b,h)
  int blk = blockIdx.x;
  int g  = blk % IT4;
  int h  = (blk / IT4) % H_;
  int b  = blk / (IT4 * H_);
  int it = g * 4 + warp;    // this wave's 16-row i-tile

  const u16* qh = q  + (((size_t)b*H_ + h)*N_ + it*16)*D_;
  const u16* kh = kk + ((size_t)b*H_ + h)*(size_t)N_*D_;
  const u16* vh = vv + ((size_t)b*H_ + h)*(size_t)D_*N_;

  v16bf qa0 = load_afrag(qh + (size_t)l16*D_ +  0 + hi*16);
  v16bf qa1 = load_afrag(qh + (size_t)l16*D_ + 32 + hi*16);

  v8f a0 = {}, a1 = {}, a2 = {}, a3 = {};
  float m[8], l[8];
#pragma unroll
  for (int r = 0; r < 8; ++r) { m[r] = -3.0e38f; l[r] = 0.f; }
  u16* pl = &Pl[warp][0];

  // per-thread async-copy slices: K tile = 256 16B chunks, V tile = 256 16B chunks
  const int kr0 = tid >> 3,         kc0 = tid & 7;          // chunks tid, tid+128
  const int kr1 = (tid + 128) >> 3, kc1 = (tid + 128) & 7;
  const int vr0 = tid >> 2,         vc0 = tid & 3;
  const int vr1 = (tid + 128) >> 2, vc1 = (tid + 128) & 3;
  const unsigned klo = lds_off(&Kl[0][0]);
  const unsigned vlo = lds_off(&Vl[0][0]);

#define ISSUE_TILE(buf, j0)                                                        \
  do {                                                                             \
    unsigned kb = klo + (unsigned)(buf) * (32 * 64 * 2);                           \
    unsigned vb2 = vlo + (unsigned)(buf) * (64 * 32 * 2);                          \
    async_ld_b128(kb + kr0 * 128 + kc0 * 16,                                       \
                  (const char*)(kh + (size_t)((j0) + kr0) * D_) + kc0 * 16);       \
    async_ld_b128(kb + kr1 * 128 + kc1 * 16,                                       \
                  (const char*)(kh + (size_t)((j0) + kr1) * D_) + kc1 * 16);       \
    async_ld_b128(vb2 + vr0 * 64 + vc0 * 16,                                       \
                  (const char*)(vh + (size_t)vr0 * N_ + (j0)) + vc0 * 16);         \
    async_ld_b128(vb2 + vr1 * 64 + vc1 * 16,                                       \
                  (const char*)(vh + (size_t)vr1 * N_ + (j0)) + vc1 * 16);         \
  } while (0)

  ISSUE_TILE(0, 0);

  for (int j0 = 0, step = 0; j0 < N_; j0 += 32, ++step) {
    int cur = step & 1;
    if (j0 + 32 < N_) {
      ISSUE_TILE(cur ^ 1, j0 + 32);
      wait_async_le();   // current tile's 4 copies complete; next 4 in flight
    } else {
      wait_async_0();
    }
    __syncthreads();     // current tile visible to all 4 waves

    const u16* Kb = &Kl[cur][0];
    const u16* Vb = &Vl[cur][0];

    // ---- S tiles (16 x 32) = q(16x64) * k^T, operands from LDS ----
    v8f s0 = {}, s1 = {};
    {
      v16bf k00 = *(const v16bf*)(Kb + ( 0 + l16)*64 +  0 + hi*16);
      v16bf k01 = *(const v16bf*)(Kb + ( 0 + l16)*64 + 32 + hi*16);
      v16bf k10 = *(const v16bf*)(Kb + (16 + l16)*64 +  0 + hi*16);
      v16bf k11 = *(const v16bf*)(Kb + (16 + l16)*64 + 32 + hi*16);
      s0 = wmma_bf16(qa0, k00, s0);
      s0 = wmma_bf16(qa1, k01, s0);
      s1 = wmma_bf16(qa0, k10, s1);
      s1 = wmma_bf16(qa1, k11, s1);
    }
    // ---- online softmax (row r lives in component r across the 16-lane group) ----
    float p0[8], p1[8];
#pragma unroll
    for (int r = 0; r < 8; ++r) {
      float mx = fmaxf(s0[r], s1[r]);
      mx = fmaxf(mx, __shfl_xor(mx, 1, 16));
      mx = fmaxf(mx, __shfl_xor(mx, 2, 16));
      mx = fmaxf(mx, __shfl_xor(mx, 4, 16));
      mx = fmaxf(mx, __shfl_xor(mx, 8, 16));
      float mnew  = fmaxf(m[r], mx);
      float alpha = __expf(m[r] - mnew);
      float e0 = __expf(s0[r] - mnew);
      float e1 = __expf(s1[r] - mnew);
      float sum = e0 + e1;
      sum += __shfl_xor(sum, 1, 16);
      sum += __shfl_xor(sum, 2, 16);
      sum += __shfl_xor(sum, 4, 16);
      sum += __shfl_xor(sum, 8, 16);
      l[r] = l[r] * alpha + sum;
      m[r] = mnew;
      a0[r] *= alpha; a1[r] *= alpha; a2[r] *= alpha; a3[r] *= alpha;
      p0[r] = e0; p1[r] = e1;
    }
    // ---- P: C-layout -> A-layout via per-wave LDS region ----
#pragma unroll
    for (int r = 0; r < 8; ++r) {
      int row = r + hi * 8;
      pl[row*32 + l16]      = f2bf(p0[r]);
      pl[row*32 + 16 + l16] = f2bf(p1[r]);
    }
    asm volatile("" ::: "memory");  // keep DS store -> DS load ordered (in-order per wave)
    v16bf pa = load_afrag(pl + l16*32 + hi*16);
    asm volatile("" ::: "memory");
    // ---- acc += P(16x32) * V(32x64), V from LDS ----
    v16bf v0 = *(const v16bf*)(Vb + (size_t)( 0 + l16)*32 + hi*16);
    v16bf v1 = *(const v16bf*)(Vb + (size_t)(16 + l16)*32 + hi*16);
    v16bf v2 = *(const v16bf*)(Vb + (size_t)(32 + l16)*32 + hi*16);
    v16bf v3 = *(const v16bf*)(Vb + (size_t)(48 + l16)*32 + hi*16);
    a0 = wmma_bf16(pa, v0, a0);
    a1 = wmma_bf16(pa, v1, a1);
    a2 = wmma_bf16(pa, v2, a2);
    a3 = wmma_bf16(pa, v3, a3);

    __syncthreads();     // all waves done reading `cur` before it is overwritten
  }
#undef ISSUE_TILE

  // ---- epilogue: divide by l, store [b][n=i][h][d] bf16 ----
  u16* orow = ao + (((size_t)b*N_ + it*16 + hi*8)*HD_) + h*D_ + l16;
#pragma unroll
  for (int r = 0; r < 8; ++r) {
    float rl = 1.0f / l[r];
    orow[(size_t)r*HD_ +  0] = f2bf(a0[r] * rl);
    orow[(size_t)r*HD_ + 16] = f2bf(a1[r] * rl);
    orow[(size_t)r*HD_ + 32] = f2bf(a2[r] * rl);
    orow[(size_t)r*HD_ + 48] = f2bf(a3[r] * rl);
  }
}

// ---------------- output GEMM: out[b][o][n] = Wout[o][hd] * ao[hd][n] + bout[o] ----------------
__global__ void k_gemm_out(const u16* __restrict__ A, const u16* __restrict__ Bm,
                           const float* __restrict__ bias, float* __restrict__ outp) {
  const int lane = threadIdx.x & 31;
  const int warp = threadIdx.x >> 5;
  const int l16 = lane & 15, hi = lane >> 4;
  int w = blockIdx.x * 4 + warp;
  const int NS = N_ / 64;  // 32
  const int MT = C_ / 16;  // 32
  int b  = w / (MT * NS);
  int r0 = w % (MT * NS);
  int ot = r0 / NS;
  int ns = r0 % NS;
  const u16* arow = A + (size_t)(ot*16 + l16)*HD_ + hi*16;
  const u16* brow = Bm + ((size_t)b*N_ + ns*64)*HD_ + hi*16;
  v8f a0 = {}, a1 = {}, a2 = {}, a3 = {};
  for (int k0 = 0; k0 < HD_; k0 += 32) {
    v16bf a  = load_afrag(arow + k0);
    v16bf b0 = *(const v16bf*)(brow + (size_t)( 0 + l16)*HD_ + k0);
    v16bf b1 = *(const v16bf*)(brow + (size_t)(16 + l16)*HD_ + k0);
    v16bf b2 = *(const v16bf*)(brow + (size_t)(32 + l16)*HD_ + k0);
    v16bf b3 = *(const v16bf*)(brow + (size_t)(48 + l16)*HD_ + k0);
    a0 = wmma_bf16(a, b0, a0);
    a1 = wmma_bf16(a, b1, a1);
    a2 = wmma_bf16(a, b2, a2);
    a3 = wmma_bf16(a, b3, a3);
  }
  float* orow = outp + ((size_t)b*C_ + ot*16 + hi*8)*N_ + ns*64 + l16;
#pragma unroll
  for (int r = 0; r < 8; ++r) {
    float bb = bias[ot*16 + r + hi*8];
    orow[(size_t)r*N_ +  0] = a0[r] + bb;
    orow[(size_t)r*N_ + 16] = a1[r] + bb;
    orow[(size_t)r*N_ + 32] = a2[r] + bb;
    orow[(size_t)r*N_ + 48] = a3[r] + bb;
  }
}

extern "C" void kernel_launch(void* const* d_in, const int* in_sizes, int n_in,
                              void* d_out, int out_size, void* d_ws, size_t ws_size,
                              hipStream_t stream) {
  const float* x    = (const float*)d_in[0];  // [B, C, N]
  const float* Wqkv = (const float*)d_in[1];  // [3072, 512]
  const float* qsc  = (const float*)d_in[2];  // [64]
  const float* ksc  = (const float*)d_in[3];  // [64]
  const float* Wout = (const float*)d_in[4];  // [512, 1024]
  const float* bout = (const float*)d_in[5];  // [512]
  float* out = (float*)d_out;                 // [B, C, N] f32

  char* ws = (char*)d_ws;
  // workspace layout (total 90 MiB)
  const size_t o_xt  = 0;                      // xT  bf16 [B][N][C]      : 4 MiB
  const size_t o_wq  = (size_t)4  << 20;       // WqkvB bf16              : 3 MiB
  const size_t o_wo  = (size_t)8  << 20;       // WoutB bf16              : 1 MiB
  const size_t o_q   = (size_t)10 << 20;       // q bf16 [B][H][N][D]     : 8 MiB
  const size_t o_k   = (size_t)18 << 20;       // k bf16 [B][H][N][D]     : 8 MiB
  const size_t o_v   = (size_t)26 << 20;       // v bf16 [B][H][D][N]     : 8 MiB
  const size_t o_ao  = (size_t)34 << 20;       // attn out bf16 [B][N][HD]: 8 MiB
  const size_t o_qkv = (size_t)42 << 20;       // qkv f32 [B][3072][N]    : 48 MiB

  u16*   xT    = (u16*)(ws + o_xt);
  u16*   WqkvB = (u16*)(ws + o_wq);
  u16*   WoutB = (u16*)(ws + o_wo);
  u16*   qb    = (u16*)(ws + o_q);
  u16*   kb    = (u16*)(ws + o_k);
  u16*   vb    = (u16*)(ws + o_v);
  u16*   aob   = (u16*)(ws + o_ao);
  float* qkv   = (float*)(ws + o_qkv);

  // 1) weight conversions
  k_cvt_bf16<<<(M3_*C_)/256, 256, 0, stream>>>(Wqkv, WqkvB, M3_*C_);
  k_cvt_bf16<<<(C_*HD_)/256, 256, 0, stream>>>(Wout, WoutB, C_*HD_);
  // 2) x transpose -> bf16
  k_transpose_x<<<B_*(C_/32)*(N_/32), 256, 0, stream>>>(x, xT);
  // 3) QKV GEMM (12288 waves)
  k_gemm_qkv<<<(B_*(M3_/16)*(N_/64))/4, 128, 0, stream>>>(WqkvB, xT, qkv);
  // 4) normalize+pack q,k ; convert v
  k_normqk<<<B_*2*H_*(N_/32), 256, 0, stream>>>(qkv, qsc, ksc, qb, kb);
  k_cvt_v<<<(B_*HD_*N_)/256, 256, 0, stream>>>(qkv, vb);
  // 5) flash attention: 1024 blocks x 4 waves, block-cooperative K/V staging
  k_flash<<<B_*H_*(N_/64), 128, 0, stream>>>(qb, kb, vb, aob);
  // 6) output projection + bias (2048 waves)
  k_gemm_out<<<(B_*(C_/16)*(N_/64))/4, 128, 0, stream>>>(WoutB, aob, bout, out);
}
